// Multi_EfficientAttention_432_56229711839614
// MI455X (gfx1250) — matrix operand order repaired and checked by reading, not compiled, not run on hardware
//
#include <hip/hip_runtime.h>
#include <hip/hip_bf16.h>
#include <stdint.h>

typedef __attribute__((ext_vector_type(16))) __bf16 v16bf;
typedef __attribute__((ext_vector_type(8)))  float  v8f;

__device__ __forceinline__ unsigned short f2bf(float f) {
  unsigned int u = __float_as_uint(f);
  u += 0x7FFFu + ((u >> 16) & 1u);   // round-to-nearest-even bf16
  return (unsigned short)(u >> 16);
}

// ---------------- elementwise (x [+ y]) -> bf16 ----------------
__global__ void addcvt_kernel(const float* __restrict__ x,
                              const float* __restrict__ y,
                              unsigned short* __restrict__ o, long n) {
  long i = (long)blockIdx.x * blockDim.x + threadIdx.x;
  long stride = (long)gridDim.x * blockDim.x;
  for (; i < n; i += stride) {
    float v = x[i];
    if (y) v += y[i];
    o[i] = f2bf(v);
  }
}

// ---------- softmax over query axis Nq, per (b, channel c) ----------
// grid.x = B * (D/64); block = 256 = 64 channels x 4 row-strides
__global__ void softmax_q_kernel(const float* __restrict__ q,
                                 unsigned short* __restrict__ qs,
                                 int B, int Nq, int D) {
  int groups = D / 64;
  int b = blockIdx.x / groups;
  int cg = blockIdx.x % groups;
  int cl = threadIdx.x & 63;
  int r0 = threadIdx.x >> 6;                 // 0..3
  int c = cg * 64 + cl;
  const float* base = q + (size_t)b * Nq * D + c;

  __shared__ float red[4][64];

  float m = -3.4e38f;
  for (int n = r0; n < Nq; n += 4) m = fmaxf(m, base[(size_t)n * D]);
  red[r0][cl] = m;
  __syncthreads();
  float cm = fmaxf(fmaxf(red[0][cl], red[1][cl]), fmaxf(red[2][cl], red[3][cl]));
  __syncthreads();

  float s = 0.f;
  for (int n = r0; n < Nq; n += 4) s += __expf(base[(size_t)n * D] - cm);
  red[r0][cl] = s;
  __syncthreads();
  float inv = 1.0f / (red[0][cl] + red[1][cl] + red[2][cl] + red[3][cl]);

  unsigned short* ob = qs + (size_t)b * Nq * D + c;
  for (int n = r0; n < Nq; n += 4)
    ob[(size_t)n * D] = f2bf(__expf(base[(size_t)n * D] - cm) * inv);
}

// ---------- softmax over 64 head-channels, one wave32 per head ----------
// grid.x = rows (B*Nk); block = 256 = 8 heads
__global__ void softmax_k_kernel(const float* __restrict__ k,
                                 unsigned short* __restrict__ ks, long rows) {
  long row = blockIdx.x;
  if (row >= rows) return;
  int h = threadIdx.x >> 5;
  int lane = threadIdx.x & 31;
  const float* p = k + row * 512 + h * 64;
  float a = p[lane], b = p[lane + 32];
  float m = fmaxf(a, b);
  for (int o = 16; o > 0; o >>= 1) m = fmaxf(m, __shfl_xor(m, o, 32));
  float e0 = __expf(a - m), e1 = __expf(b - m);
  float s = e0 + e1;
  for (int o = 16; o > 0; o >>= 1) s += __shfl_xor(s, o, 32);
  float inv = 1.0f / s;
  unsigned short* ob = ks + row * 512 + h * 64;
  ob[lane]      = f2bf(e0 * inv);
  ob[lane + 32] = f2bf(e1 * inv);
}

// ---------------- generic batched bf16 WMMA GEMM ----------------
// C[m,n] = sum_k A[m*Asr + k*Asc] * W[n*Wsr + k*Wsc] (+ bias[n])
// batches: blockIdx.z -> (bb = z/nh, hh = z%nh); per-batch element offsets.
// output: fp32 (Cf) or bf16 (Cb), ldc row stride.
#define LDT 48   // LDS row stride (elems): 96B = 16B-aligned, conflict-padded

__global__ void __launch_bounds__(256)
wmma_gemm_bf16(const unsigned short* __restrict__ A,
               const unsigned short* __restrict__ W,
               const float* __restrict__ bias,
               float* __restrict__ Cf,
               unsigned short* __restrict__ Cb,
               int M, int N, int K,
               long Asr, long Asc, long Wsr, long Wsc, long ldc,
               int nh, long Ab, long Ah, long Wb, long Wh, long Cbs, long Chs) {
  __shared__ __align__(16) unsigned short As[128 * LDT];
  __shared__ __align__(16) unsigned short Bs[128 * LDT];

  int z = blockIdx.z;
  int bb = z / nh, hh = z % nh;
  const unsigned short* Ap = A + (size_t)bb * Ab + (size_t)hh * Ah;
  const unsigned short* Wp = W + (size_t)bb * Wb + (size_t)hh * Wh;
  long coff = (long)bb * Cbs + (long)hh * Chs;

  int n0 = blockIdx.x * 128;
  int m0 = blockIdx.y * 128;
  int tid = threadIdx.x;
  int lane = tid & 31;
  int wave = tid >> 5;
  int wm = wave & 3, wn = wave >> 2;   // 4x2 wave grid; wave tile 32(M)x64(N)
  int lr = lane & 15;
  int hf = lane >> 4;

  v8f acc[2][4];
  for (int j = 0; j < 4; j++) {
    int gn = n0 + 64 * wn + 16 * j + lr;
    float bv = (bias && gn < N) ? bias[gn] : 0.0f;
    for (int i = 0; i < 2; i++)
      for (int e = 0; e < 8; e++) acc[i][j][e] = bv;
  }

  const bool fastA = (Asc == 1);
  const bool fastW = (Wsc == 1);

  for (int k0 = 0; k0 < K; k0 += 32) {
    // ---- stage A tile (128 x 32) ----
    if (fastA) {
      for (int c = tid; c < 512; c += 256) {
        int r = c >> 2, c4 = c & 3;
        int gm = m0 + r, gk = k0 + c4 * 8;
        uint4 val = make_uint4(0, 0, 0, 0);
        if (gm < M && gk + 7 < K) {
          val = *reinterpret_cast<const uint4*>(Ap + (size_t)gm * Asr + gk);
          if (k0 + 32 < K)
            __builtin_prefetch(Ap + (size_t)gm * Asr + gk + 32, 0, 0);
        }
        *reinterpret_cast<uint4*>(&As[r * LDT + c4 * 8]) = val;
      }
    } else {
      for (int c = tid; c < 4096; c += 256) {
        int r = c >> 5, kk = c & 31;
        int gm = m0 + r, gk = k0 + kk;
        unsigned short v = 0;
        if (gm < M && gk < K) v = Ap[(size_t)gm * Asr + (size_t)gk * Asc];
        As[r * LDT + kk] = v;
      }
    }
    // ---- stage W tile (128 x 32) ----
    if (fastW) {
      for (int c = tid; c < 512; c += 256) {
        int r = c >> 2, c4 = c & 3;
        int gn = n0 + r, gk = k0 + c4 * 8;
        uint4 val = make_uint4(0, 0, 0, 0);
        if (gn < N && gk + 7 < K)
          val = *reinterpret_cast<const uint4*>(Wp + (size_t)gn * Wsr + gk);
        *reinterpret_cast<uint4*>(&Bs[r * LDT + c4 * 8]) = val;
      }
    } else {
      for (int c = tid; c < 4096; c += 256) {
        int r = c >> 5, kk = c & 31;
        int gn = n0 + r, gk = k0 + kk;
        unsigned short v = 0;
        if (gn < N && gk < K) v = Wp[(size_t)gn * Wsr + (size_t)gk * Wsc];
        Bs[r * LDT + kk] = v;
      }
    }
    __syncthreads();

    // ---- fragment gathers per documented gfx1250 16-bit layouts ----
    // A 16x32: lane(M=lr,hf): VGPR0-3 K = hf*8+0..7, VGPR4-7 K = 16+hf*8+0..7
    // B 32x16: lane(N=lr,hf): VGPR0-3 K = hf*8+0..7, VGPR4-7 K = 16+hf*8+0..7
    union Frag { v16bf v; uint4 q[2]; };
    Frag af[2], bfr[4];
    for (int i = 0; i < 2; i++) {
      const unsigned short* p = &As[(32 * wm + 16 * i + lr) * LDT];
      af[i].q[0] = *reinterpret_cast<const uint4*>(p + hf * 8);
      af[i].q[1] = *reinterpret_cast<const uint4*>(p + 16 + hf * 8);
    }
    for (int j = 0; j < 4; j++) {
      const unsigned short* p = &Bs[(64 * wn + 16 * j + lr) * LDT];
      bfr[j].q[0] = *reinterpret_cast<const uint4*>(p + hf * 8);
      bfr[j].q[1] = *reinterpret_cast<const uint4*>(p + 16 + hf * 8);
    }
    for (int i = 0; i < 2; i++)
      for (int j = 0; j < 4; j++)
        acc[i][j] = __builtin_amdgcn_wmma_f32_16x16x32_bf16(
            false, af[i].v, false, bfr[j].v, (short)0, acc[i][j], false, false);
    __syncthreads();
  }

  // ---- store C (32-bit C/D layout: VGPR v -> row v + 8*hf, col lr) ----
  for (int i = 0; i < 2; i++)
    for (int j = 0; j < 4; j++) {
      int gn = n0 + 64 * wn + 16 * j + lr;
      if (gn >= N) continue;
      for (int v = 0; v < 8; v++) {
        int gm = m0 + 32 * wm + 16 * i + v + 8 * hf;
        if (gm >= M) continue;
        long idx = coff + (long)gm * ldc + gn;
        if (Cf) Cf[idx] = acc[i][j][v];
        else    Cb[idx] = f2bf(acc[i][j][v]);
      }
    }
}

extern "C" void kernel_launch(void* const* d_in, const int* in_sizes, int n_in,
                              void* d_out, int out_size, void* d_ws, size_t ws_size,
                              hipStream_t stream) {
  (void)in_sizes; (void)n_in; (void)out_size; (void)ws_size;
  const float* f2  = (const float*)d_in[0];
  const float* f3  = (const float*)d_in[1];
  const float* f4  = (const float*)d_in[2];
  const float* f2p = (const float*)d_in[3];
  const float* f3p = (const float*)d_in[4];
  const float* f4p = (const float*)d_in[5];
  const float* Wq1 = (const float*)d_in[6];  const float* bq1 = (const float*)d_in[7];
  const float* Wk1 = (const float*)d_in[8];  const float* bk1 = (const float*)d_in[9];
  const float* Wv1 = (const float*)d_in[10]; const float* bv1 = (const float*)d_in[11];
  const float* Wq2 = (const float*)d_in[12]; const float* bq2 = (const float*)d_in[13];
  const float* Wk2 = (const float*)d_in[14]; const float* bk2 = (const float*)d_in[15];
  const float* Wv2 = (const float*)d_in[16]; const float* bv2 = (const float*)d_in[17];
  const float* Wr  = (const float*)d_in[18]; const float* br  = (const float*)d_in[19];
  float* out = (float*)d_out;

  const int B = 8, Nq = 4096, N3 = 1024, N4 = 256, D = 512, H = 8;
  const long Mq = (long)B * Nq, M3 = (long)B * N3, M4 = (long)B * N4;

  char* wp = (char*)d_ws;
  auto carve = [&](size_t bytes) -> void* {
    void* p = (void*)wp; wp += (bytes + 255) & ~(size_t)255; return p;
  };
  unsigned short* f2b  = (unsigned short*)carve(Mq * D * 2);
  unsigned short* f3pb = (unsigned short*)carve(M3 * D * 2);
  unsigned short* f3vb = (unsigned short*)carve(M3 * D * 2);
  unsigned short* f4pb = (unsigned short*)carve(M4 * D * 2);
  unsigned short* f4vb = (unsigned short*)carve(M4 * D * 2);
  unsigned short* wq1b = (unsigned short*)carve((size_t)D * D * 2);
  unsigned short* wk1b = (unsigned short*)carve((size_t)D * D * 2);
  unsigned short* wv1b = (unsigned short*)carve((size_t)D * D * 2);
  unsigned short* wq2b = (unsigned short*)carve((size_t)D * D * 2);
  unsigned short* wk2b = (unsigned short*)carve((size_t)D * D * 2);
  unsigned short* wv2b = (unsigned short*)carve((size_t)D * D * 2);
  unsigned short* wrb  = (unsigned short*)carve((size_t)D * 2 * D * 2);
  float*          scr  = (float*)carve(Mq * D * 4);          // reused fp32 scratch
  unsigned short* qs1  = (unsigned short*)carve(Mq * D * 2);
  unsigned short* qs2  = (unsigned short*)carve(Mq * D * 2);
  unsigned short* ks3  = (unsigned short*)carve(M3 * D * 2);
  unsigned short* v3b  = (unsigned short*)carve(M3 * D * 2);
  unsigned short* ks4  = (unsigned short*)carve(M4 * D * 2);
  unsigned short* v4b  = (unsigned short*)carve(M4 * D * 2);
  unsigned short* ctx1 = (unsigned short*)carve((size_t)B * H * 64 * 64 * 2);
  unsigned short* ctx2 = (unsigned short*)carve((size_t)B * H * 64 * 64 * 2);
  unsigned short* cat  = (unsigned short*)carve(Mq * 2 * D * 2);

  auto cvt = [&](const float* x, const float* y, unsigned short* o, long n) {
    long blocks = (n + 255) / 256; if (blocks > 8192) blocks = 8192;
    addcvt_kernel<<<dim3((unsigned)blocks), dim3(256), 0, stream>>>(x, y, o, n);
  };
  auto gemm = [&](const unsigned short* A, const unsigned short* W,
                  const float* bias, float* Cf, unsigned short* Cb,
                  int M, int N, int K,
                  long Asr, long Asc, long Wsr, long Wsc, long ldc,
                  int nb, int nh, long Ab, long Ah, long Wb_, long Wh,
                  long Cbs, long Chs) {
    dim3 g((N + 127) / 128, (M + 127) / 128, nb * nh);
    wmma_gemm_bf16<<<g, dim3(256), 0, stream>>>(A, W, bias, Cf, Cb, M, N, K,
        Asr, Asc, Wsr, Wsc, ldc, nh, Ab, Ah, Wb_, Wh, Cbs, Chs);
  };

  // 1) weights + activations -> bf16 (pos-embed adds fused)
  cvt(Wq1, nullptr, wq1b, (long)D * D);
  cvt(Wk1, nullptr, wk1b, (long)D * D);
  cvt(Wv1, nullptr, wv1b, (long)D * D);
  cvt(Wq2, nullptr, wq2b, (long)D * D);
  cvt(Wk2, nullptr, wk2b, (long)D * D);
  cvt(Wv2, nullptr, wv2b, (long)D * D);
  cvt(Wr,  nullptr, wrb,  (long)D * 2 * D);
  cvt(f2, f2p, f2b,  Mq * D);
  cvt(f3, f3p, f3pb, M3 * D);
  cvt(f3, nullptr, f3vb, M3 * D);
  cvt(f4, f4p, f4pb, M4 * D);
  cvt(f4, nullptr, f4vb, M4 * D);

  // 2) projections + softmaxes
  gemm(f2b, wq1b, bq1, scr, nullptr, (int)Mq, D, D, D, 1, D, 1, D, 1, 1, 0,0,0,0,0,0);
  softmax_q_kernel<<<dim3(B * (D / 64)), dim3(256), 0, stream>>>(scr, qs1, B, Nq, D);
  gemm(f2b, wq2b, bq2, scr, nullptr, (int)Mq, D, D, D, 1, D, 1, D, 1, 1, 0,0,0,0,0,0);
  softmax_q_kernel<<<dim3(B * (D / 64)), dim3(256), 0, stream>>>(scr, qs2, B, Nq, D);

  gemm(f3pb, wk1b, bk1, scr, nullptr, (int)M3, D, D, D, 1, D, 1, D, 1, 1, 0,0,0,0,0,0);
  softmax_k_kernel<<<dim3((unsigned)M3), dim3(256), 0, stream>>>(scr, ks3, M3);
  gemm(f3vb, wv1b, bv1, nullptr, v3b, (int)M3, D, D, D, 1, D, 1, D, 1, 1, 0,0,0,0,0,0);

  gemm(f4pb, wk2b, bk2, scr, nullptr, (int)M4, D, D, D, 1, D, 1, D, 1, 1, 0,0,0,0,0,0);
  softmax_k_kernel<<<dim3((unsigned)M4), dim3(256), 0, stream>>>(scr, ks4, M4);
  gemm(f4vb, wv2b, bv2, nullptr, v4b, (int)M4, D, D, D, 1, D, 1, D, 1, 1, 0,0,0,0,0,0);

  // 3) context[b,h,d,e] = sum_n ks[b,n,h,d] * v[b,n,h,e]   (M=d, N=e, K=n)
  gemm(ks3, v3b, nullptr, nullptr, ctx1, 64, 64, N3,
       /*Asr*/1, /*Asc*/D, /*Wsr*/1, /*Wsc*/D, /*ldc*/64,
       B, H, (long)N3 * D, 64, (long)N3 * D, 64, (long)H * 4096, 4096);
  gemm(ks4, v4b, nullptr, nullptr, ctx2, 64, 64, N4,
       1, D, 1, D, 64, B, H, (long)N4 * D, 64, (long)N4 * D, 64, (long)H * 4096, 4096);

  // 4) attended[b,n,h,e] = sum_d qs[b,n,h,d] * ctx[b,h,d,e] -> concat halves
  gemm(qs1, ctx1, nullptr, nullptr, cat,       Nq, 64, 64,
       /*Asr*/D, /*Asc*/1, /*Wsr*/1, /*Wsc*/64, /*ldc*/2 * D,
       B, H, (long)Nq * D, 64, (long)H * 4096, 4096, (long)Nq * 2 * D, 64);
  gemm(qs2, ctx2, nullptr, nullptr, cat + D,   Nq, 64, 64,
       D, 1, 1, 64, 2 * D,
       B, H, (long)Nq * D, 64, (long)H * 4096, 4096, (long)Nq * 2 * D, 64);

  // 5) out = concat @ Wr^T + br  (fp32 output)
  gemm(cat, wrb, br, out, nullptr, (int)Mq, D, 2 * D,
       2 * D, 1, 2 * D, 1, D, 1, 1, 0, 0, 0, 0, 0, 0);
}
